// Conv2DLayer_29308856828627
// MI455X (gfx1250) — compile-verified, matrix operand order checked by reading, not run on hardware
//
#include <hip/hip_runtime.h>
#include <cmath>

// ---------------------------------------------------------------------------
// Fused upfirdn2d(up=2, fir=[1,3,3,1]^2) + 3x3 conv (true conv) + bias/lrelu/clamp
//
// Polyphase fusion: zero-insert-up2 -> FIR4x4 -> conv3x3 composes into a 6x6
// kernel on the zero-inserted grid; by output parity (rm,rn) this splits into
// four 3x3 kernels applied directly to x.  Implemented as implicit GEMM on
// V_WMMA_F32_16X16X32_F16.
//
// Round-3 change: each wave owns TWO 16-column M-tiles (32 x-columns), so each
// LDS B-fragment (1 KB/wave) feeds two WMMAs -> halves ds_load traffic per
// WMMA (LDS B-fetch was the projected WGP bottleneck at ~1KB/clk demand vs
// ~256B/clk supply).  Block = 8 waves = 2 input rows x 128 columns.
// ---------------------------------------------------------------------------

typedef __attribute__((ext_vector_type(16))) _Float16 v16h;
typedef __attribute__((ext_vector_type(8)))  _Float16 v8h;
typedef __attribute__((ext_vector_type(8)))  float    v8f;

union ABFrag { v16h v; v8h h[2]; };

#define NIMG   16
#define ICH    64
#define OCH    64
#define HH     128
#define WW     128
#define XP_H   130
#define XP_W   130

// halfs per phase of packed weights: 9 taps * 64 k * 64 n
#define WB_PHASE_HALFS (9 * 64 * 64)

// ---------------------------------------------------------------------------
// Kernel 1: NCHW f32 -> padded NHWC f16  (xp[n][h+1][w+1][ic], zero border)
// ---------------------------------------------------------------------------
__global__ void pack_x_kernel(const float* __restrict__ x, _Float16* __restrict__ xp)
{
    int idx = blockIdx.x * blockDim.x + threadIdx.x;
    const int total = NIMG * XP_H * XP_W;
    if (idx >= total) return;
    int wp = idx % XP_W;
    int t  = idx / XP_W;
    int hp = t % XP_H;
    int n  = t / XP_H;
    _Float16* dst = xp + (size_t)idx * ICH;
    const int hh = hp - 1, ww = wp - 1;
    if (hh >= 0 && hh < HH && ww >= 0 && ww < WW) {
        const float* sp = x + (((size_t)n * ICH) * HH + hh) * WW + ww;
        #pragma unroll 8
        for (int ic = 0; ic < ICH; ++ic)
            dst[ic] = (_Float16)sp[(size_t)ic * (HH * WW)];
    } else {
        #pragma unroll 8
        for (int ic = 0; ic < ICH; ++ic)
            dst[ic] = (_Float16)0.0f;
    }
}

// ---------------------------------------------------------------------------
// Kernel 2: build polyphase 3x3 kernels, store f16 in WMMA-B fragment order:
//   Wb[ ((((ph*9 + tap)*2 + kb)*4 + nb)*32 + lane)*16 + j ],  ph = rm*2 + rn
// where k = ic (kb=ic/32), n = oc (nb=oc/16), (lane,j) per 16-bit fragment map.
// ---------------------------------------------------------------------------
__global__ void pack_w_kernel(const float* __restrict__ W, const float* __restrict__ f,
                              _Float16* __restrict__ Wb)
{
    int idx = blockIdx.x * blockDim.x + threadIdx.x;
    if (idx >= OCH * ICH) return;
    const int ic = idx & 63;
    const int oc = idx >> 6;
    const float g = 1.0f / 24.0f;  // 1/sqrt(64*9)

    // wf[c][d2] = g * weight[oc][ic][2-c][2-d2]   (flipped -> true convolution)
    float wf[3][3];
    #pragma unroll
    for (int c = 0; c < 3; ++c)
        #pragma unroll
        for (int d2 = 0; d2 < 3; ++d2)
            wf[c][d2] = g * W[(((size_t)oc * ICH + ic) * 3 + (2 - c)) * 3 + (2 - d2)];

    float f4[4][4];
    #pragma unroll
    for (int a = 0; a < 4; ++a)
        #pragma unroll
        for (int b = 0; b < 4; ++b)
            f4[a][b] = 4.0f * f[a * 4 + b];   // gain = up^2

    // Kc = full correlation of wf with f4 -> 6x6 composed kernel
    float Kc[6][6];
    #pragma unroll
    for (int p = 0; p < 6; ++p) {
        #pragma unroll
        for (int q = 0; q < 6; ++q) {
            float s = 0.0f;
            #pragma unroll
            for (int c = 0; c < 3; ++c) {
                const int a = p - c;
                if (a < 0 || a > 3) continue;
                #pragma unroll
                for (int d2 = 0; d2 < 3; ++d2) {
                    const int b = q - d2;
                    if (b < 0 || b > 3) continue;
                    s += wf[c][d2] * f4[a][b];
                }
            }
            Kc[p][q] = s;
        }
    }

    const int kb = ic >> 5, kr = ic & 31;
    const int nb = oc >> 4, nl = oc & 15;
    // 16-bit fragment mapping: (kr,nl) -> (lane, half index j)
    int l, j;
    if      (kr < 8)  { l = nl;      j = kr;      }
    else if (kr < 16) { l = nl + 16; j = kr - 8;  }
    else if (kr < 24) { l = nl;      j = kr - 8;  }
    else              { l = nl + 16; j = kr - 16; }

    #pragma unroll
    for (int ph = 0; ph < 4; ++ph) {
        const int rm = ph >> 1, rn = ph & 1;
        #pragma unroll
        for (int d = 0; d < 3; ++d) {
            #pragma unroll
            for (int e = 0; e < 3; ++e) {
                const float val = Kc[2 * d + 1 - rm][2 * e + 1 - rn];
                const int tap = d * 3 + e;
                const size_t o =
                    ((size_t)((((ph * 9 + tap) * 2 + kb) * 4 + nb) * 32 + l)) * 16 + j;
                Wb[o] = (_Float16)val;
            }
        }
    }
}

// ---------------------------------------------------------------------------
// Kernel 3: main WMMA conv.  Block = (image n, input row pair h0) x rm parity.
// Waves 0-3 -> row h0, waves 4-7 -> row h0+1; each wave: 32 x-columns
// (two M-tiles), all 64 oc, both rn phases.  288 WMMAs / wave;
// each A fragment feeds 8 WMMAs, each B fragment feeds 2 WMMAs.
// ---------------------------------------------------------------------------
__global__ __launch_bounds__(256) void conv2d_up2_wmma(
    const _Float16* __restrict__ xp, const _Float16* __restrict__ Wb,
    const float* __restrict__ bias, float* __restrict__ out)
{
    extern __shared__ _Float16 wlds[];   // 2 phases * 9*64*64 halfs = 144 KB

    const int rm  = blockIdx.y;          // output row parity
    const int n   = blockIdx.x >> 6;
    const int h0  = (blockIdx.x & 63) << 1;
    const int tid = threadIdx.x;

    // Stage weights for phases (rm,0) and (rm,1): 147456 B = 36 uint4/thread.
    {
        const uint4* src = (const uint4*)(Wb + (size_t)(rm * 2) * WB_PHASE_HALFS);
        uint4* dst = (uint4*)wlds;
        #pragma unroll
        for (int i = 0; i < 36; ++i)
            dst[tid + i * 256] = src[tid + i * 256];
    }
    __syncthreads();

    const int lane    = tid & 31;
    const int wave    = tid >> 5;
    const int row     = h0 + (wave >> 2);    // input row for this wave
    const int w0      = (wave & 3) << 5;     // 32 x-columns per wave
    const int lp      = lane & 15;           // M position / N column within tile
    const int hiHalf  = lane >> 4;
    const int koffLn  = hiHalf << 3;         // 16-bit A-fragment lane K offset

    if (h0 < 126)  // prefetch next row pair (global_prefetch_b8)
        __builtin_prefetch(xp + (size_t)(n * XP_H + h0 + 4) * XP_W * ICH, 0, 0);

    v8f acc[2][2][4] = {};   // [M tile][rn phase][oc tile], f32 accumulate

    #pragma unroll
    for (int d = 0; d < 3; ++d) {
        const _Float16* rowp = xp + (size_t)(n * XP_H + row + d) * XP_W * ICH;
        #pragma unroll
        for (int e = 0; e < 3; ++e) {
            const int t = d * 3 + e;
            const _Float16* posp0 = rowp + (size_t)(w0 + lp + e) * ICH;
            const _Float16* posp1 = posp0 + 16 * ICH;
            #pragma unroll
            for (int kb = 0; kb < 2; ++kb) {
                ABFrag a0, a1;
                const int ko = (kb << 5) + koffLn;
                a0.h[0] = *(const v8h*)(posp0 + ko);
                a0.h[1] = *(const v8h*)(posp0 + ko + 16);
                a1.h[0] = *(const v8h*)(posp1 + ko);
                a1.h[1] = *(const v8h*)(posp1 + ko + 16);
                #pragma unroll
                for (int p = 0; p < 2; ++p) {           // rn parity
                    const _Float16* bbase =
                        wlds + (size_t)((((p * 9 + t) * 2 + kb) * 4) * 32 + lane) * 16;
                    #pragma unroll
                    for (int nb = 0; nb < 4; ++nb) {    // oc tile
                        ABFrag b;
                        const _Float16* bp = bbase + nb * 512;  // 32 lanes*16 halfs
                        b.h[0] = *(const v8h*)(bp);
                        b.h[1] = *(const v8h*)(bp + 8);
                        acc[0][p][nb] = __builtin_amdgcn_wmma_f32_16x16x32_f16(
                            false, a0.v, false, b.v, (short)0, acc[0][p][nb],
                            false, false);
                        acc[1][p][nb] = __builtin_amdgcn_wmma_f32_16x16x32_f16(
                            false, a1.v, false, b.v, (short)0, acc[1][p][nb],
                            false, false);
                    }
                }
            }
        }
    }

    // Epilogue: bias + lrelu(0.2)*sqrt(2) + clamp(+-256).
    // Both rn parities -> contiguous float2 per position; per lane per oc-tile
    // per M-tile the 8 float2 form one 64B-aligned fully-covered segment.
    const int orow = 2 * row + rm;
    const float SQ2 = 1.4142135623730951f;
    #pragma unroll
    for (int mt = 0; mt < 2; ++mt) {
        const int wbase = w0 + (mt << 4);
        #pragma unroll
        for (int nb = 0; nb < 4; ++nb) {
            const int oc = (nb << 4) + lp;
            const float bv = bias[oc];
            float* op = out + ((size_t)(n * OCH + oc) * 256 + orow) * 256;
            #pragma unroll
            for (int r = 0; r < 8; ++r) {
                const int m = r + (hiHalf << 3);   // spatial index within tile
                float v0 = acc[mt][0][nb][r] + bv;
                v0 = (v0 >= 0.0f ? v0 : 0.2f * v0) * SQ2;
                v0 = fminf(fmaxf(v0, -256.0f), 256.0f);
                float v1 = acc[mt][1][nb][r] + bv;
                v1 = (v1 >= 0.0f ? v1 : 0.2f * v1) * SQ2;
                v1 = fminf(fmaxf(v1, -256.0f), 256.0f);
                float2 pair = make_float2(v0, v1);
                *(float2*)(op + (size_t)2 * (wbase + m)) = pair;
            }
        }
    }
}

// ---------------------------------------------------------------------------
extern "C" void kernel_launch(void* const* d_in, const int* in_sizes, int n_in,
                              void* d_out, int out_size, void* d_ws, size_t ws_size,
                              hipStream_t stream)
{
    (void)in_sizes; (void)n_in; (void)out_size; (void)ws_size;
    const float* x      = (const float*)d_in[0];
    const float* weight = (const float*)d_in[1];
    const float* bias   = (const float*)d_in[2];
    const float* f      = (const float*)d_in[3];
    float* out          = (float*)d_out;

    // Workspace layout: xp (padded NHWC f16, 34,611,200 B) then Wb (294,912 B).
    _Float16* xp = (_Float16*)d_ws;
    const size_t xpBytes = (size_t)NIMG * XP_H * XP_W * ICH * sizeof(_Float16);
    _Float16* Wb = (_Float16*)((char*)d_ws + xpBytes);

    {
        const int total = NIMG * XP_H * XP_W;
        pack_x_kernel<<<(total + 255) / 256, 256, 0, stream>>>(x, xp);
    }
    pack_w_kernel<<<(OCH * ICH + 255) / 256, 256, 0, stream>>>(weight, f, Wb);

    dim3 grid(NIMG * (HH / 2), 2);   // (n, row pair) x rm parity
    conv2d_up2_wmma<<<grid, 256,
                      2 * WB_PHASE_HALFS * sizeof(_Float16), stream>>>(
        xp, Wb, bias, out);
}
